// MTGCNNet_14491219656878
// MI455X (gfx1250) — matrix operand order repaired and checked by reading, not compiled, not run on hardware
//
#include <hip/hip_runtime.h>
#include <hip/hip_bf16.h>
#include <math.h>

#define HID 100
#define NPAD 112   // 7*16 padded output cols
#define KPAD 128   // padded K (max K = 58+58 = 116)

typedef _Float16 v16h __attribute__((ext_vector_type(16)));
typedef _Float16 v8h  __attribute__((ext_vector_type(8)));
typedef float    v8f  __attribute__((ext_vector_type(8)));

// ---------------- degree / norm ----------------
__global__ void deg_kernel(const int* __restrict__ row, const int* __restrict__ col,
                           float* __restrict__ deg, int e) {
    int i = blockIdx.x * blockDim.x + threadIdx.x;
    if (i < e) {
        int r = row[i];
        if (r != col[i]) atomicAdd(&deg[r], 1.0f);
    }
}

__global__ void dinv_kernel(const float* __restrict__ deg, float* __restrict__ dinv, int n) {
    int i = blockIdx.x * blockDim.x + threadIdx.x;
    if (i < n) {
        float d = deg[i];
        dinv[i] = (d > 0.0f) ? rsqrtf(fmaxf(d, 1.0f)) : 0.0f;
    }
}

// ------------- conv1 aggregation: tx1[col] += nw * x[row]  (wave per edge) -------------
__global__ void tx1_scatter_kernel(const int* __restrict__ row, const int* __restrict__ col,
                                   const float* __restrict__ dinv, const float* __restrict__ x,
                                   float* __restrict__ tx1, int e) {
    int gt = blockIdx.x * blockDim.x + threadIdx.x;
    int wid = gt >> 5, lane = gt & 31;
    if (wid >= e) return;
    int r = row[wid], c = col[wid];
    if (r == c) return;
    float nw = -dinv[r] * dinv[c];
    if (nw == 0.0f) return;
    const float* xr = x + (long)r * 58;
    float* td = tx1 + (long)c * 58;
    if (lane < 58) atomicAdd(&td[lane], nw * xr[lane]);
    int l2 = lane + 32;
    if (l2 < 58) atomicAdd(&td[l2], nw * xr[l2]);
}

// ------------- WMMA GEMM: out = [relu((A0|A1) @ W + bias)] (+ addvec broadcast) -------------
// A0: [n,k0], A1: [n,k1] (optional), W: [(k0+k1) x HID] row-major, out: [n,HID]
__global__ __launch_bounds__(224) void gemm_wmma_kernel(
    const float* __restrict__ A0, int k0,
    const float* __restrict__ A1, int k1,
    const float* __restrict__ W,
    const float* __restrict__ bias,
    const float* __restrict__ addvec,   // may be null; broadcast per-row add AFTER relu
    float* __restrict__ out, int n)
{
    __shared__ __attribute__((aligned(32))) _Float16 sA[16][KPAD];
    __shared__ __attribute__((aligned(32))) _Float16 sB[NPAD][KPAD];   // K-transposed: sB[n][k]
    const int tid  = threadIdx.x;
    const int tile = blockIdx.x;
    const int K = k0 + k1;

    // stage A tile (f32 -> f16, zero-padded)
    for (int idx = tid; idx < 16 * KPAD; idx += 224) {
        int m = idx >> 7, k = idx & (KPAD - 1);
        int r = tile * 16 + m;
        float v = 0.0f;
        if (r < n) {
            if (k < k0)      v = A0[(long)r * k0 + k];
            else if (k < K)  v = A1[(long)r * k1 + (k - k0)];
        }
        sA[m][k] = (_Float16)v;
    }
    // stage B tile transposed (f32 -> f16, zero-padded)
    for (int idx = tid; idx < NPAD * KPAD; idx += 224) {
        int nn = idx >> 7, k = idx & (KPAD - 1);
        float v = (k < K && nn < HID) ? W[(long)k * HID + nn] : 0.0f;
        sB[nn][k] = (_Float16)v;
    }
    __syncthreads();

    const int wave = tid >> 5;        // 0..6 -> 16-col tile
    const int lane = tid & 31;
    const int half = lane >> 4;
    const int l15  = lane & 15;
    const int nbase = wave * 16;

    v8f c = {};
    #pragma unroll
    for (int kk = 0; kk < KPAD; kk += 32) {
        // A operand layout (contiguous, matches B):
        // elem j -> K = kk + 16*half + j ; M = l15
        union { v16h v; struct { v8h lo, hi; } s; } ua;
        ua.s.lo = *(const v8h*)&sA[l15][kk + 16 * half];
        ua.s.hi = *(const v8h*)&sA[l15][kk + 16 * half + 8];
        // B operand: elem j -> K = kk + 16*half + j ; N = l15
        v16h b = *(const v16h*)&sB[nbase + l15][kk + 16 * half];
        c = __builtin_amdgcn_wmma_f32_16x16x32_f16(false, ua.v, false, b,
                                                   (short)0, c, false, false);
    }

    // D layout: c[r] -> M = r + 8*half, N = l15
    const int colo = nbase + l15;
    #pragma unroll
    for (int r = 0; r < 8; ++r) {
        int ro = tile * 16 + r + 8 * half;
        if (ro < n && colo < HID) {
            float v = c[r] + bias[colo];
            v = v > 0.0f ? v : 0.0f;
            if (addvec) v += addvec[ro];
            out[(long)ro * HID + colo] = v;
        }
    }
}

// ------------- conv2 projections: x1_base = h@W2[0]+b2 ; p1 = h@W2[1]  (wave per node) -------------
__global__ void p_kernel(const float* __restrict__ h, const float* __restrict__ W2,
                         const float* __restrict__ b2, float* __restrict__ x1,
                         float* __restrict__ p1, int n) {
    int gt = blockIdx.x * blockDim.x + threadIdx.x;
    int node = gt >> 5, lane = gt & 31;
    if (node >= n) return;
    const float* hr = h + (long)node * HID;
    float s0 = 0.0f, s1 = 0.0f;
    for (int k = lane; k < HID; k += 32) {
        float hv = hr[k];
        s0 += hv * W2[k];
        s1 += hv * W2[HID + k];
    }
    for (int off = 16; off > 0; off >>= 1) {
        s0 += __shfl_down(s0, off);
        s1 += __shfl_down(s1, off);
    }
    if (lane == 0) {
        x1[node] = s0 + b2[0];
        p1[node] = s1;
    }
}

// ------------- conv2 aggregation (reordered to scalar scatter): x1[col] += nw * p1[row] -------------
__global__ void x1_scatter_kernel(const int* __restrict__ row, const int* __restrict__ col,
                                  const float* __restrict__ dinv, const float* __restrict__ p1,
                                  float* __restrict__ x1, int e) {
    int i = blockIdx.x * blockDim.x + threadIdx.x;
    if (i < e) {
        int r = row[i], c = col[i];
        if (r != c) {
            float nw = -dinv[r] * dinv[c];
            if (nw != 0.0f) atomicAdd(&x1[c], nw * p1[r]);
        }
    }
}

// ------------- link-pred loss: wave per (pos|neg) edge -------------
__global__ void loss_kernel(const int* __restrict__ ei, const int* __restrict__ ni,
                            const float* __restrict__ z, float* __restrict__ acc, int e) {
    __shared__ float red[2];
    if (threadIdx.x == 0) { red[0] = 0.0f; red[1] = 0.0f; }
    __syncthreads();
    int gt = blockIdx.x * blockDim.x + threadIdx.x;
    int wid = gt >> 5, lane = gt & 31;
    if (wid < 2 * e) {
        int isNeg = (wid >= e) ? 1 : 0;
        int ee = isNeg ? wid - e : wid;
        const int* idx = isNeg ? ni : ei;
        int a = idx[ee], b = idx[e + ee];
        const float* za = z + (long)a * HID;
        const float* zb = z + (long)b * HID;
        float dot = 0.0f;
        for (int k = lane; k < HID; k += 32) dot += za[k] * zb[k];
        for (int off = 16; off > 0; off >>= 1) dot += __shfl_down(dot, off);
        if (lane == 0) {
            float sig = 1.0f / (1.0f + expf(-dot));
            float t = isNeg ? logf(1.0f - sig + 1e-15f) : logf(sig + 1e-15f);
            atomicAdd(&red[isNeg], t);
        }
    }
    __syncthreads();
    if (threadIdx.x == 0) {
        atomicAdd(&acc[0], red[0]);
        atomicAdd(&acc[1], red[1]);
    }
}

__global__ void finalize_kernel(const float* __restrict__ acc, const float* __restrict__ c1,
                                const float* __restrict__ c2, float* __restrict__ out,
                                int n, float invE) {
    out[n]     = -(acc[0] + acc[1]) * invE;
    out[n + 1] = c1[0];
    out[n + 2] = c2[0];
}

extern "C" void kernel_launch(void* const* d_in, const int* in_sizes, int n_in,
                              void* d_out, int out_size, void* d_ws, size_t ws_size,
                              hipStream_t stream) {
    const float* x      = (const float*)d_in[0];
    const int*   ei     = (const int*)d_in[1];
    const int*   ni     = (const int*)d_in[2];
    const float* W1     = (const float*)d_in[3];   // [2,58,100] == [116,100] flat
    const float* b1     = (const float*)d_in[4];
    const float* W2     = (const float*)d_in[5];   // [2,100,1]  == [200] flat
    const float* b2     = (const float*)d_in[6];
    const float* lin2_W = (const float*)d_in[9];
    const float* lin2_b = (const float*)d_in[10];
    const float* c1     = (const float*)d_in[11];
    const float* c2     = (const float*)d_in[12];

    const int N = in_sizes[0] / 58;
    const int E = in_sizes[1] / 2;
    float* out = (float*)d_out;

    // workspace layout (floats): [deg N][tx1 58N][acc 2][dinv N][p1 N][h 100N][z 100N]
    float* ws   = (float*)d_ws;
    float* deg  = ws;
    float* tx1  = deg + N;
    float* acc  = tx1 + (size_t)58 * N;
    float* dinv = acc + 2;
    float* p1   = dinv + N;
    float* h    = p1 + N;
    float* z    = h + (size_t)100 * N;

    // zero all accumulated-into scratch each call (graph-replay safe)
    hipMemsetAsync(ws, 0, ((size_t)59 * N + 2) * sizeof(float), stream);

    const int* row = ei;       // edge_index[0]
    const int* col = ei + E;   // edge_index[1]

    deg_kernel<<<(E + 255) / 256, 256, 0, stream>>>(row, col, deg, E);
    dinv_kernel<<<(N + 255) / 256, 256, 0, stream>>>(deg, dinv, N);
    // wave(32) per edge
    tx1_scatter_kernel<<<(E + 7) / 8, 256, 0, stream>>>(row, col, dinv, x, tx1, E);
    // h = relu([x|tx1] @ W1(116x100) + b1)
    gemm_wmma_kernel<<<(N + 15) / 16, 224, 0, stream>>>(x, 58, tx1, 58, W1, b1, nullptr, h, N);
    // x1 base + p1
    p_kernel<<<(N + 7) / 8, 256, 0, stream>>>(h, W2, b2, out, p1, N);
    // x1 += segment_sum(nw * p1[row] -> col)
    x1_scatter_kernel<<<(E + 255) / 256, 256, 0, stream>>>(row, col, dinv, p1, out, E);
    // z = x1 + relu(x @ lin2_W + lin2_b)
    gemm_wmma_kernel<<<(N + 15) / 16, 224, 0, stream>>>(x, 58, nullptr, 0, lin2_W, lin2_b, out, z, N);
    // pos+neg loss over 2E edges, wave per edge
    loss_kernel<<<(2 * E + 7) / 8, 256, 0, stream>>>(ei, ni, z, acc, E);
    finalize_kernel<<<1, 1, 0, stream>>>(acc, c1, c2, out, N, 1.0f / (float)E);
}
